// FeatureFusion_6803228197124
// MI455X (gfx1250) — matrix operand order repaired and checked by reading, not compiled, not run on hardware
//
#include <hip/hip_runtime.h>
#include <hip/hip_bf16.h>

// =====================================================================
// Fully-fused FeatureFusion forward for MI455X (gfx1250).
//  - one block = 8 samples, entire network computed in LDS
//  - all K>=32 linears via v_wmma_f32_16x16x32_bf16 (f32 accumulate)
//  - A and W staged once per GEMM as bf16 in bank-padded LDS tiles;
//    fragments loaded as 2x ds_load_b128 per operand (no inner-loop cvt)
//  - bias staged to LDS; branch-free epilogue (outputs padded to 16 rows)
//  - per-sample attentions / tiny heads via scalar VALU (negligible FLOPs)
// =====================================================================

typedef __bf16 bf16_t;
typedef __attribute__((ext_vector_type(16))) __bf16 v16bf;
typedef __attribute__((ext_vector_type(8)))  __bf16 v8bf;
typedef __attribute__((ext_vector_type(8)))  float  v8f;

#define NT 256          // threads per block (8 wave32 waves)
#define NW 8            // waves per block
#define S  8            // samples per block
#define D  128          // model dim
#define KPAD 8          // bf16 row pad (16B) -> conflict-free fragment loads

// ---- parameter leaf indices (insertion-order flattening of setup dict) ----
// 0: x
// parts: 6 x 12 leaves  [w,b,g,be, rw1,rb1,rg1,rbe1, rw2,rb2,rg2,rbe2]
#define IPART(p) (1 + (p) * 12)
// mha: 73..76 [in_w,in_b,out_w,out_b]
// msa scales: 3 x 24 leaves starting at 77:
//  [fp_w,fp_b,fp_g,fp_be, ca_w1,ca_b1,ca_g,ca_be, ca_w2,ca_b2,
//   sa_w1,sa_b1,sa_g,sa_be, sa_w2,sa_b2, qkv_w,qkv_b,
//   fe_w1,fe_b1,fe_g,fe_be, fe_w2,fe_b2]
#define ISCALE(bi,k) (77 + (bi) * 24 + (k))
// 149 fw_w, 150 fw_b, 151 proj_w, 152 proj_b
// 153..156 dyn_w,dyn_b,dyn_g,dyn_be
// 157..162 enh [w1,b1,g,be,w2,b2]
// 163..168 cal [w1,b1,g,be,w2,b2]
// 169..176 agg_res [w1,b1,g1,be1,w2,b2,g2,be2]
// 177 agg_g, 178 agg_be, 179 out_g, 180 out_be
#define NPAR 181

struct PP { const float* p[NPAR]; };

// ---- LDS layout (floats then bf16), sizes in elements ----
#define F_XS    (S * 52)          // input slice [8][52]
#define F_ABUF  (48 * D)          // [48][128]
#define F_WIDE  (48 * 384)        // [48][384]
#define F_PM    (16 * D)          // pooled [8][128], padded to 16 rows (GEMM out)
#define F_C32   (16 * 32)         // [8][32], padded to 16 rows (GEMM out)
#define F_BMEAN (3 * S * D)       // branch means [3][8][128]
#define F_WF    (S * 4)           // fusion weights
#define F_DWT   (S * 8)           // dyn weights
#define F_SWV   (64)              // spatial gate per row
#define F_BSTG  (128)             // staged bias chunk
#define SMEM_FLOATS (F_XS + 3 * F_ABUF + F_WIDE + F_PM + F_C32 + F_BMEAN + F_WF + F_DWT + F_SWV + F_BSTG)
#define B_ABF   (48  * (256 + KPAD))   // bf16 A staging  [48][K<=256 (+pad)]
#define B_WBFT  (128 * (256 + KPAD))   // bf16 W staging, transposed [NB<=128][K<=256 (+pad)]
#define B_BR    (3 * S * 6 * D)        // bf16 branches
#define SMEM_BYTES (SMEM_FLOATS * 4 + (B_ABF + B_WBFT + B_BR) * 2)

__device__ __forceinline__ float gelu_f(float x) {
  return 0.5f * x * (1.0f + erff(x * 0.7071067811865475f));
}
__device__ __forceinline__ float sigm_f(float x) {
  return 1.0f / (1.0f + __expf(-x));
}

// ---------------------------------------------------------------------
// Cooperative WMMA GEMM:  out[M][NB] = A_lds[M][K] @ W[:,ncol0:ncol0+NB] + bias
// A is f32 in LDS (lda), W is f32 in global (row-major K x Nfull).
// Both staged to bf16: abf row-major [M][K+KPAD]; wbfT transposed
// [NB][K+KPAD] so each lane's 16 fragment values are contiguous in K.
// Bias chunk staged to LDS -> branch-free epilogue.
// K in {32,128,256}, NB in {32,128} (powers of two; shifts not divides).
// All out targets have >= ceil16(Mrows) rows of writable LDS: partial
// M-tiles store garbage rows into padding (never read).
// reuseA: skip A staging (same A/K as previous call, e.g. N-chunked GEMMs).
// ---------------------------------------------------------------------
__device__ void gemm_wmma(const float* __restrict__ W, int Nfull, int ncol0, int NB,
                          const float* __restrict__ bias,
                          const float* A_lds, int lda, int Mrows, int K,
                          float* out_lds, int ldo, int ocol0,
                          bf16_t* abf, bf16_t* wbfT, float* bstage,
                          int tid, bool reuseA = false)
{
  const int ldab   = K + KPAD;
  const int kshift = __builtin_ctz(K);
  const int nshift = __builtin_ctz(NB);
  __syncthreads();
  if (!reuseA) {
    for (int idx = tid; idx < Mrows * K; idx += NT) {
      int r = idx >> kshift, k = idx & (K - 1);
      abf[(size_t)r * ldab + k] = (bf16_t)A_lds[(size_t)r * lda + k];
    }
  }
  for (int idx = tid; idx < K * NB; idx += NT) {
    int k = idx >> nshift, n = idx & (NB - 1);      // coalesced global read
    wbfT[(size_t)n * ldab + k] = (bf16_t)W[(size_t)k * Nfull + ncol0 + n];
  }
  if (tid < NB) bstage[tid] = bias[ncol0 + tid];
  __syncthreads();

  int lane = tid & 31, wid = tid >> 5;
  int rr = lane & 15;
  int khalfA = (lane >> 4) << 3;   // A frag: +0 or +8   (K pattern {0..7,16..23}/{8..15,24..31})
  int kaddB  = (lane >> 4) << 3;   // B frag: +0 or +8   (K pattern {0..7,16..23}/{8..15,24..31})
  int Mt = (Mrows + 15) >> 4, Ntl = NB >> 4;
  int tshift = __builtin_ctz(Ntl);

  for (int t = wid; t < Mt * Ntl; t += NW) {
    int mi = t >> tshift, ni = t & (Ntl - 1);
    v8f acc = {0.f, 0.f, 0.f, 0.f, 0.f, 0.f, 0.f, 0.f};
    const bf16_t* arow = abf  + (size_t)(mi * 16 + rr) * ldab + khalfA;
    const bf16_t* brow = wbfT + (size_t)(ni * 16 + rr) * ldab + kaddB;
    for (int k0 = 0; k0 < K; k0 += 32) {
      v8bf a0 = *(const v8bf*)(arow + k0);
      v8bf a1 = *(const v8bf*)(arow + k0 + 16);
      v8bf b0 = *(const v8bf*)(brow + k0);
      v8bf b1 = *(const v8bf*)(brow + k0 + 16);
      v16bf a = __builtin_shufflevector(a0, a1, 0,1,2,3,4,5,6,7,8,9,10,11,12,13,14,15);
      v16bf b = __builtin_shufflevector(b0, b1, 0,1,2,3,4,5,6,7,8,9,10,11,12,13,14,15);
      acc = __builtin_amdgcn_wmma_f32_16x16x32_bf16(false, a, false, b,
                                                    (short)0, acc, false, false);
    }
    int col  = ni * 16 + rr;
    float bv = bstage[col];
    float* op = out_lds + (size_t)(mi * 16 + ((lane < 16) ? 0 : 8)) * ldo + ocol0 + col;
#pragma unroll
    for (int e = 0; e < 8; ++e) op[(size_t)e * ldo] = acc[e] + bv;
  }
  __syncthreads();
}

// Row-wise LayerNorm over C cols (+optional act: 1=gelu, 2=relu), wave per row.
__device__ void ln_rows(float* buf, int ld, int rows, int C,
                        const float* g, const float* be, int act, int tid)
{
  int lane = tid & 31, wid = tid >> 5;
  for (int r = wid; r < rows; r += NW) {
    float* row = buf + (size_t)r * ld;
    float s = 0.f, s2 = 0.f;
    for (int c = lane; c < C; c += 32) { float v = row[c]; s += v; s2 += v * v; }
#pragma unroll
    for (int off = 16; off > 0; off >>= 1) {
      s  += __shfl_xor(s,  off, 32);
      s2 += __shfl_xor(s2, off, 32);
    }
    float m   = s / (float)C;
    float var = fmaxf(s2 / (float)C - m * m, 0.f);
    float inv = rsqrtf(var + 1e-5f);
    for (int c = lane; c < C; c += 32) {
      float v = (row[c] - m) * inv * g[c] + be[c];
      if (act == 1) v = gelu_f(v);
      else if (act == 2) v = fmaxf(v, 0.f);
      row[c] = v;
    }
  }
  __syncthreads();
}

// torch F.interpolate(mode='linear', align_corners=False) along token axis.
__device__ void interp_tok(const float* src, int ldi, int Hi,
                           float* dst, int ldo, int Ho, int tid)
{
  for (int idx = tid; idx < S * Ho * D; idx += NT) {
    int c = idx & (D - 1);
    int t = (idx >> 7) % Ho;
    int s = idx / (D * Ho);
    float pos = ((float)t + 0.5f) * ((float)Hi / (float)Ho) - 0.5f;
    pos = fminf(fmaxf(pos, 0.f), (float)(Hi - 1));
    int i0 = (int)floorf(pos);
    int i1 = (i0 + 1 < Hi) ? i0 + 1 : Hi - 1;
    float w = pos - (float)i0;
    dst[(size_t)(s * Ho + t) * ldo + c] =
        src[(size_t)(s * Hi + i0) * ldi + c] * (1.f - w) +
        src[(size_t)(s * Hi + i1) * ldi + c] * w;
  }
  __syncthreads();
}

__global__ __launch_bounds__(NT)
void ff_fused_kernel(const float* __restrict__ x, float* __restrict__ out,
                     int Btot, PP pp)
{
  extern __shared__ char smem[];
  float* xs    = (float*)smem;            // [8][52]
  float* Ab    = xs + F_XS;               // [48][128]
  float* Bb    = Ab + F_ABUF;             // [48][128]
  float* Cc    = Bb + F_ABUF;             // [48][128]
  float* wide  = Cc + F_ABUF;             // [48][384]
  float* pm    = wide + F_WIDE;           // [8][128] (+8 pad rows)
  float* c32   = pm + F_PM;               // [8][32]  (+8 pad rows)
  float* bmean = c32 + F_C32;             // [3][8][128]
  float* wf    = bmean + F_BMEAN;         // [8][4]
  float* dwt   = wf + F_WF;               // [8][8]
  float* swv   = dwt + F_DWT;             // [64]
  float* bstage= swv + F_SWV;             // [128] staged bias
  bf16_t* abf  = (bf16_t*)(bstage + F_BSTG); // [48][K+pad]
  bf16_t* wbfT = abf + B_ABF;             // [NB][K+pad] transposed weights
  bf16_t* br   = wbfT + B_WBFT;           // [3][8][6][128]

  int tid  = threadIdx.x;
  int lane = tid & 31, wid = tid >> 5;
  int b0   = blockIdx.x * S;

  // ---- load input slice ----
  for (int idx = tid; idx < S * 50; idx += NT) {
    int s = idx / 50, c = idx - s * 50;
    xs[s * 52 + c] = (b0 + s < Btot) ? x[(size_t)(b0 + s) * 50 + c] : 0.f;
  }
  __syncthreads();

  // =================== Stage 1: per-part feature extractors ===================
  const int sp_off[6] = {0, 9, 17, 25, 33, 41};
  const int sp_len[6] = {9, 8, 8, 8, 8, 9};
  for (int p = 0; p < 6; ++p) {
    // h = x_slice @ W (K<=9 -> scalar), rows = 8
    const float* w = pp.p[IPART(p) + 0];
    const float* b = pp.p[IPART(p) + 1];
    for (int idx = tid; idx < S * D; idx += NT) {
      int n = idx & 127, s = idx >> 7;
      float acc = b[n];
      const float* ip = xs + s * 52 + sp_off[p];
      for (int k = 0; k < sp_len[p]; ++k) acc += ip[k] * w[(size_t)k * D + n];
      Bb[idx] = acc;
    }
    __syncthreads();
    ln_rows(Bb, D, S, D, pp.p[IPART(p) + 2], pp.p[IPART(p) + 3], 1, tid); // LN+GELU
    // resblock
    gemm_wmma(pp.p[IPART(p) + 4], D, 0, D, pp.p[IPART(p) + 5], Bb, D, S, D, Cc, D, 0,
              abf, wbfT, bstage, tid);
    ln_rows(Cc, D, S, D, pp.p[IPART(p) + 6], pp.p[IPART(p) + 7], 2, tid);  // LN+ReLU
    gemm_wmma(pp.p[IPART(p) + 8], D, 0, D, pp.p[IPART(p) + 9], Cc, D, S, D, wide, 384, 0,
              abf, wbfT, bstage, tid);
    ln_rows(wide, 384, S, D, pp.p[IPART(p) + 10], pp.p[IPART(p) + 11], 0, tid);
    for (int idx = tid; idx < S * D; idx += NT) {
      int c = idx & 127, s = idx >> 7;
      Ab[(size_t)(s * 6 + p) * D + c] = Bb[idx] + wide[(size_t)s * 384 + c];   // feats
    }
    __syncthreads();
  }

  // =================== Stage 2: MHA over N=6 tokens (nh=4,hd=32) ===================
  for (int c = 0; c < 3; ++c)
    gemm_wmma(pp.p[73], 384, c * 128, 128, pp.p[74], Ab, D, 48, D, wide, 384, c * 128,
              abf, wbfT, bstage, tid, c > 0);
  {
    int s = wid;                       // one wave per sample
    int h = lane >> 3, n = lane & 7;   // 4 heads x 8 slots
    if (n < 6) {
      const float* base = wide + (size_t)s * 6 * 384;
      float sc[6], mx = -1e30f;
      for (int m = 0; m < 6; ++m) {
        float d2 = 0.f;
        const float* qv = base + (size_t)n * 384 + h * 32;
        const float* kv = base + (size_t)m * 384 + 128 + h * 32;
        for (int e = 0; e < 32; ++e) d2 += qv[e] * kv[e];
        sc[m] = d2 * 0.17677669529663687f;
        mx = fmaxf(mx, sc[m]);
      }
      float sum = 0.f;
      for (int m = 0; m < 6; ++m) { sc[m] = __expf(sc[m] - mx); sum += sc[m]; }
      float inv = 1.f / sum;
      for (int e = 0; e < 32; ++e) {
        float o = 0.f;
        for (int m = 0; m < 6; ++m) o += sc[m] * base[(size_t)m * 384 + 256 + h * 32 + e];
        Cc[(size_t)(s * 6 + n) * D + h * 32 + e] = o * inv;
      }
    }
  }
  __syncthreads();
  gemm_wmma(pp.p[75], D, 0, D, pp.p[76], Cc, D, 48, D, Bb, D, 0,
            abf, wbfT, bstage, tid);  // att -> Bb

  // =================== Stage 3: Multi-Scale Attention (3 branches) ===================
  const int HsArr[3] = {6, 4, 4};
  for (int bi = 0; bi < 3; ++bi) {
    int Hs = HsArr[bi];
    int rows = S * Hs;
    interp_tok(Bb, D, 6, Ab, D, Hs, tid);                               // xs_tok -> Ab
    gemm_wmma(pp.p[ISCALE(bi,0)], D, 0, D, pp.p[ISCALE(bi,1)], Ab, D, rows, D, Cc, D, 0,
              abf, wbfT, bstage, tid);
    ln_rows(Cc, D, rows, D, pp.p[ISCALE(bi,2)], pp.p[ISCALE(bi,3)], 1, tid);
    // channel gate: seq-mean -> 2-layer MLP -> sigmoid
    for (int idx = tid; idx < S * D; idx += NT) {
      int c = idx & 127, s = idx >> 7;
      float m = 0.f;
      for (int t = 0; t < Hs; ++t) m += Cc[(size_t)(s * Hs + t) * D + c];
      pm[idx] = m / (float)Hs;
    }
    __syncthreads();
    gemm_wmma(pp.p[ISCALE(bi,4)], 32, 0, 32, pp.p[ISCALE(bi,5)], pm, D, S, D, c32, 32, 0,
              abf, wbfT, bstage, tid);
    ln_rows(c32, 32, S, 32, pp.p[ISCALE(bi,6)], pp.p[ISCALE(bi,7)], 1, tid);
    gemm_wmma(pp.p[ISCALE(bi,8)], D, 0, D, pp.p[ISCALE(bi,9)], c32, 32, S, 32, pm, D, 0,
              abf, wbfT, bstage, tid);
    for (int idx = tid; idx < rows * D; idx += NT) {
      int c = idx & 127, s = (idx >> 7) / Hs;
      Cc[idx] *= sigm_f(pm[s * D + c]);
    }
    __syncthreads();
    // spatial gate
    gemm_wmma(pp.p[ISCALE(bi,10)], 32, 0, 32, pp.p[ISCALE(bi,11)], Cc, D, rows, D,
              wide, 384, 0, abf, wbfT, bstage, tid);
    ln_rows(wide, 384, rows, 32, pp.p[ISCALE(bi,12)], pp.p[ISCALE(bi,13)], 1, tid);
    if (tid < rows) {
      const float* w2 = pp.p[ISCALE(bi,14)];
      float sacc = pp.p[ISCALE(bi,15)][0];
      for (int k = 0; k < 32; ++k) sacc += wide[(size_t)tid * 384 + k] * w2[k];
      swv[tid] = sigm_f(sacc);
    }
    __syncthreads();
    for (int idx = tid; idx < rows * D; idx += NT) Cc[idx] *= swv[idx >> 7];
    __syncthreads();
    // qkv + biased attention (nh=8, hd=16)
    for (int c = 0; c < 3; ++c)
      gemm_wmma(pp.p[ISCALE(bi,16)], 384, c * 128, 128, pp.p[ISCALE(bi,17)], Cc, D, rows, D,
                wide, 384, c * 128, abf, wbfT, bstage, tid, c > 0);
    {
      int s = wid;
      for (int u = lane; u < 8 * Hs; u += 32) {
        int h = u / Hs, n = u - h * Hs;
        const float* base = wide + (size_t)s * Hs * 384;
        float sc[6], mx = -1e30f;
        for (int m = 0; m < Hs; ++m) {
          float d2 = 0.f;
          const float* qv = base + (size_t)n * 384 + h * 16;
          const float* kv = base + (size_t)m * 384 + 128 + h * 16;
          for (int e = 0; e < 16; ++e) d2 += qv[e] * kv[e];
          sc[m] = d2 * 0.25f + (float)(n - m) * 0.02f;
          mx = fmaxf(mx, sc[m]);
        }
        float sum = 0.f;
        for (int m = 0; m < Hs; ++m) { sc[m] = __expf(sc[m] - mx); sum += sc[m]; }
        float inv = 1.f / sum;
        for (int e = 0; e < 16; ++e) {
          float o = 0.f;
          for (int m = 0; m < Hs; ++m) o += sc[m] * base[(size_t)m * 384 + 256 + h * 16 + e];
          Ab[(size_t)(s * Hs + n) * D + h * 16 + e] = o * inv;
        }
      }
    }
    __syncthreads();
    // feature-expand MLP
    for (int c = 0; c < 2; ++c)
      gemm_wmma(pp.p[ISCALE(bi,18)], 256, c * 128, 128, pp.p[ISCALE(bi,19)], Ab, D, rows, D,
                wide, 384, c * 128, abf, wbfT, bstage, tid, c > 0);
    ln_rows(wide, 384, rows, 256, pp.p[ISCALE(bi,20)], pp.p[ISCALE(bi,21)], 1, tid);
    gemm_wmma(pp.p[ISCALE(bi,22)], D, 0, D, pp.p[ISCALE(bi,23)], wide, 384, rows, 256,
              Cc, D, 0, abf, wbfT, bstage, tid);
    // interp back to 6 tokens, store as bf16 branch
    for (int idx = tid; idx < S * 6 * D; idx += NT) {
      int c = idx & 127, t = (idx >> 7) % 6, s = idx / (6 * D);
      float pos = ((float)t + 0.5f) * ((float)Hs / 6.f) - 0.5f;
      pos = fminf(fmaxf(pos, 0.f), (float)(Hs - 1));
      int i0 = (int)floorf(pos);
      int i1 = (i0 + 1 < Hs) ? i0 + 1 : Hs - 1;
      float w = pos - (float)i0;
      float v = Cc[(size_t)(s * Hs + i0) * D + c] * (1.f - w) +
                Cc[(size_t)(s * Hs + i1) * D + c] * w;
      br[(size_t)bi * (S * 6 * D) + idx] = (bf16_t)v;
    }
    __syncthreads();
    for (int idx = tid; idx < S * D; idx += NT) {
      int c = idx & 127, s = idx >> 7;
      float m = 0.f;
      for (int t = 0; t < 6; ++t)
        m += (float)br[(size_t)bi * (S * 6 * D) + (size_t)(s * 6 + t) * D + c];
      bmean[bi * S * D + idx] = m * (1.f / 6.f);
    }
    __syncthreads();
  }
  // fusion weights: softmax(fw(concat of branch means))
  if (tid < S * 3) {
    int j = tid % 3, s = tid / 3;
    const float* fww = pp.p[149];
    float acc = pp.p[150][j];
    for (int i = 0; i < 3; ++i)
      for (int c = 0; c < D; ++c)
        acc += bmean[i * S * D + s * D + c] * fww[(size_t)(i * D + c) * 3 + j];
    wf[s * 4 + j] = acc;
  }
  __syncthreads();
  if (tid < S) {
    float* l = wf + tid * 4;
    float m = fmaxf(l[0], fmaxf(l[1], l[2]));
    float e0 = __expf(l[0] - m), e1 = __expf(l[1] - m), e2 = __expf(l[2] - m);
    float inv = 1.f / (e0 + e1 + e2);
    l[0] = e0 * inv; l[1] = e1 * inv; l[2] = e2 * inv;
  }
  __syncthreads();
  for (int idx = tid; idx < 48 * D; idx += NT) {
    int s = idx / (6 * D);
    float v = 0.f;
    for (int i = 0; i < 3; ++i)
      v += wf[s * 4 + i] * (float)br[(size_t)i * (S * 6 * D) + idx];
    Cc[idx] = v;
  }
  __syncthreads();
  gemm_wmma(pp.p[151], D, 0, D, pp.p[152], Cc, D, 48, D, Ab, D, 0,
            abf, wbfT, bstage, tid);
  for (int idx = tid; idx < 48 * D; idx += NT) Ab[idx] += Bb[idx];   // inter = proj + att
  __syncthreads();

  // =================== Stage 4: dynamic weighting + head ===================
  if (tid < 48) {
    int j = tid % 6, s = tid / 6;
    const float* dwv = pp.p[153];
    float acc = pp.p[154][j];
    const float* ip = Ab + (size_t)s * 6 * D;     // 768 contiguous (token-major)
    for (int k = 0; k < 768; ++k) acc += ip[k] * dwv[(size_t)k * 6 + j];
    dwt[s * 8 + j] = acc;
  }
  __syncthreads();
  if (tid < S) {
    float* l = dwt + tid * 8;
    float m = 0.f; for (int j = 0; j < 6; ++j) m += l[j]; m *= (1.f / 6.f);
    float v = 0.f; for (int j = 0; j < 6; ++j) { float d2 = l[j] - m; v += d2 * d2; }
    float inv = rsqrtf(v * (1.f / 6.f) + 1e-5f);
    const float* g = pp.p[155]; const float* be = pp.p[156];
    float mx = -1e30f;
    for (int j = 0; j < 6; ++j) { l[j] = (l[j] - m) * inv * g[j] + be[j]; mx = fmaxf(mx, l[j]); }
    float sum = 0.f;
    for (int j = 0; j < 6; ++j) { l[j] = __expf(l[j] - mx); sum += l[j]; }
    for (int j = 0; j < 6; ++j) l[j] /= sum;
  }
  __syncthreads();
  for (int idx = tid; idx < S * D; idx += NT) {     // weighted -> Bb[0..7]
    int c = idx & 127, s = idx >> 7;
    float v = 0.f;
    for (int n = 0; n < 6; ++n) v += Ab[(size_t)(s * 6 + n) * D + c] * dwt[s * 8 + n];
    Bb[idx] = v;
  }
  __syncthreads();
  // enhancement MLP
  for (int c = 0; c < 2; ++c)
    gemm_wmma(pp.p[157], 256, c * 128, 128, pp.p[158], Bb, D, S, D, wide, 384, c * 128,
              abf, wbfT, bstage, tid, c > 0);
  ln_rows(wide, 384, S, 256, pp.p[159], pp.p[160], 1, tid);
  gemm_wmma(pp.p[161], D, 0, D, pp.p[162], wide, 384, S, 256, Cc, D, 0,
            abf, wbfT, bstage, tid);  // enh -> Cc
  // calibration gate on concat(weighted, enh)
  for (int idx = tid; idx < S * 256; idx += NT) {
    int c = idx & 255, s = idx >> 8;
    wide[(size_t)s * 384 + c] = (c < 128) ? Bb[s * D + c] : Cc[s * D + (c - 128)];
  }
  __syncthreads();
  gemm_wmma(pp.p[163], D, 0, D, pp.p[164], wide, 384, S, 256, pm, D, 0,
            abf, wbfT, bstage, tid);
  ln_rows(pm, D, S, D, pp.p[165], pp.p[166], 1, tid);
  gemm_wmma(pp.p[167], D, 0, D, pp.p[168], pm, D, S, D, Ab, D, 0,
            abf, wbfT, bstage, tid);  // cal logits -> Ab[0..7]
  for (int idx = tid; idx < S * D; idx += NT) Cc[idx] *= sigm_f(Ab[idx]);  // out = enh*sigmoid(cal)
  __syncthreads();
  // aggregation resblock + double LN
  gemm_wmma(pp.p[169], D, 0, D, pp.p[170], Cc, D, S, D, Bb, D, 0,
            abf, wbfT, bstage, tid);
  ln_rows(Bb, D, S, D, pp.p[171], pp.p[172], 2, tid);
  gemm_wmma(pp.p[173], D, 0, D, pp.p[174], Bb, D, S, D, Ab, D, 0,
            abf, wbfT, bstage, tid);
  ln_rows(Ab, D, S, D, pp.p[175], pp.p[176], 0, tid);
  for (int idx = tid; idx < S * D; idx += NT) Cc[idx] += Ab[idx];
  __syncthreads();
  ln_rows(Cc, D, S, D, pp.p[177], pp.p[178], 0, tid);
  ln_rows(Cc, D, S, D, pp.p[179], pp.p[180], 0, tid);

  for (int idx = tid; idx < S * D; idx += NT) {
    int s = idx >> 7;
    if (b0 + s < Btot) out[(size_t)(b0 + s) * D + (idx & 127)] = Cc[idx];
  }
}

extern "C" void kernel_launch(void* const* d_in, const int* in_sizes, int n_in,
                              void* d_out, int out_size, void* d_ws, size_t ws_size,
                              hipStream_t stream)
{
  (void)d_ws; (void)ws_size; (void)out_size;
  PP pp;
  for (int i = 0; i < NPAR; ++i) pp.p[i] = (i < n_in) ? (const float*)d_in[i] : nullptr;
  int Btot = in_sizes[0] / 50;
  int grid = (Btot + S - 1) / S;
  ff_fused_kernel<<<dim3(grid), dim3(NT), SMEM_BYTES, stream>>>(
      (const float*)d_in[0], (float*)d_out, Btot, pp);
}
